// SpikingLayer_32392643346854
// MI455X (gfx1250) — matrix operand order repaired and checked by reading, not compiled, not run on hardware
//
#include <hip/hip_runtime.h>
#include <hip/hip_bf16.h>
#include <stdint.h>

#define BB 16
#define SS 2048
#define II 512
#define HH 512
#define FH 2048   // 4*H

typedef __attribute__((ext_vector_type(16))) __bf16 bf16x16;
typedef __attribute__((ext_vector_type(8)))  float  f32x8;
typedef __attribute__((ext_vector_type(4)))  unsigned int u32x4;

union FragBF { bf16x16 v; u32x4 q[2]; };

__device__ __forceinline__ unsigned short f2bf(float f) {
  unsigned int x = __float_as_uint(f);
  unsigned int r = (x + 0x7fffu + ((x >> 16) & 1u)) >> 16;  // RNE
  return (unsigned short)r;
}
__device__ __forceinline__ float bf2f(unsigned short u) {
  return __uint_as_float(((unsigned int)u) << 16);
}
__device__ __forceinline__ float sigf(float x) { return 1.0f / (1.0f + __expf(-x)); }

// A operand (16x32 bf16, row-major [m][k]): lane m=lane&15 holds K {0..7,16..23} (lo) / {8..15,24..31} (hi)
__device__ __forceinline__ bf16x16 load_a(const unsigned short* base, int ld, int lane) {
  FragBF f;
  const int row  = lane & 15;
  const int koff = (lane >= 16) ? 8 : 0;
  const unsigned short* p = base + (size_t)row * ld + koff;
  f.q[0] = *(const u32x4*)(p);
  f.q[1] = *(const u32x4*)(p + 16);
  return f.v;
}
// B operand (32x16 bf16). B[k][n] = W[n][k] with W stored [N][K] row-major:
// lane n=lane&15 holds K {0..7,16..23} (lo lanes) / {8..15,24..31} (hi lanes).
__device__ __forceinline__ bf16x16 load_b(const unsigned short* base, int ld, int lane) {
  FragBF f;
  const int col  = lane & 15;
  const int koff = (lane >= 16) ? 8 : 0;
  const unsigned short* p = base + (size_t)col * ld + koff;
  f.q[0] = *(const u32x4*)(p);
  f.q[1] = *(const u32x4*)(p + 16);
  return f.v;
}
__device__ __forceinline__ f32x8 wmma_bf16(bf16x16 a, bf16x16 b, f32x8 c) {
  return __builtin_amdgcn_wmma_f32_16x16x32_bf16(false, a, false, b, (short)0, c, false, false);
}

// CDNA5 async memory->LDS copy (GLOBAL_LOAD_ASYNC_TO_LDS_B128, tracked by ASYNCcnt).
// VDST = per-lane LDS byte offset, VADDR = per-lane 64-bit global address (GV mode).
__device__ __forceinline__ void async_load_b128_to_lds(void* lds_ptr, const void* gaddr) {
  const unsigned int lds_off = (unsigned int)(uintptr_t)lds_ptr;  // generic LDS addr: low 32 bits = LDS offset
  asm volatile("global_load_async_to_lds_b128 %0, %1, off"
               :: "v"(lds_off), "v"((unsigned long long)(uintptr_t)gaddr)
               : "memory");
}
__device__ __forceinline__ void wait_asynccnt0() {
#if __has_builtin(__builtin_amdgcn_s_wait_asynccnt)
  __builtin_amdgcn_s_wait_asynccnt(0);
#else
  asm volatile("s_wait_asynccnt 0" ::: "memory");
#endif
}

__device__ __forceinline__ void grid_barrier(int* bar, int target) {
  __syncthreads();
  if (threadIdx.x == 0) {
    __threadfence();
    atomicAdd(bar, 1);
    volatile int* vb = (volatile int*)bar;
    while (*vb < target) { __builtin_amdgcn_s_sleep(1); }
    __threadfence();
  }
  __syncthreads();
}

// ---- conversion kernels -------------------------------------------------
__global__ __launch_bounds__(256) void k_f32_to_bf16(const float* __restrict__ s,
                                                     unsigned short* __restrict__ d, int n) {
  for (int i = blockIdx.x * 256 + threadIdx.x; i < n; i += gridDim.x * 256) d[i] = f2bf(s[i]);
}
// x [B][S][I] fp32 -> xT bf16 with row r = s*16 + b  (so each timestep is one 16-row M-tile)
__global__ __launch_bounds__(256) void k_conv_xT(const float* __restrict__ x,
                                                 unsigned short* __restrict__ xT) {
  const int N = BB * SS * II;
  for (int i = blockIdx.x * 256 + threadIdx.x; i < N; i += gridDim.x * 256) {
    const int b   = i / (SS * II);
    const int rem = i - b * (SS * II);
    const int s   = rem / II;
    const int ii  = rem - s * II;
    xT[((size_t)s * 16 + b) * II + ii] = f2bf(x[i]);
  }
}

// ---- Gx = x @ W_ih^T + (b_ih + b_hh), bf16 out, layout [s*16+b][4H] ------
__global__ __launch_bounds__(256) void k_gemm_x(const unsigned short* __restrict__ xT,
                                                const unsigned short* __restrict__ Wih,
                                                const float* __restrict__ b_ih,
                                                const float* __restrict__ b_hh,
                                                unsigned short* __restrict__ Gx) {
  const int lane = threadIdx.x & 31, wave = threadIdx.x >> 5;
  const int id = blockIdx.x * 8 + wave;
  const int s  = id >> 7;
  const int n  = (id & 127) * 16;
  const unsigned short* Abase = xT + (size_t)s * 16 * II;
  const unsigned short* Bbase = Wih + (size_t)n * II;
  f32x8 acc = {};
#pragma unroll 4
  for (int kk = 0; kk < II / 32; ++kk) {
    acc = wmma_bf16(load_a(Abase + kk * 32, II, lane),
                    load_b(Bbase + kk * 32, II, lane), acc);
  }
  const int col = lane & 15, mb = (lane >= 16) ? 8 : 0;
  const float bias = b_ih[n + col] + b_hh[n + col];
#pragma unroll
  for (int j = 0; j < 8; ++j)
    Gx[((size_t)s * 16 + j + mb) * FH + n + col] = f2bf(acc[j] + bias);
}

// ---- persistent sequential sLSTM scan: 16 WGs, one per WGP --------------
__global__ __launch_bounds__(256) void k_scan(const unsigned short* __restrict__ Gx,
                                              const unsigned short* __restrict__ Whh,
                                              const float* __restrict__ thr,
                                              unsigned short* __restrict__ spks,
                                              unsigned short* __restrict__ memping,
                                              int* __restrict__ bar) {
  extern __shared__ char smem[];
  unsigned short* whh = (unsigned short*)smem;                              // 128 rows x (512+8 pad)
  float* gbuf = (float*)(smem + 128 * 520 * sizeof(unsigned short));        // [4][16][32]
  float* syn  = gbuf + 4 * 16 * 32;                                         // [16][32]
  const int w = blockIdx.x;          // h-slice owner: h in [w*32, w*32+32)
  const int tid = threadIdx.x;
  const int lane = tid & 31, wave = tid >> 5;
  const int g = wave >> 1, c = wave & 1;       // gate, column-subtile
  const int ncol = lane & 15, mb = (lane >= 16) ? 8 : 0;
  const float th = thr[0];

  // Stage W_hh slice into LDS once via async global->LDS DMA (ASYNCcnt path):
  // local row lr -> global row (lr/32)*512 + w*32 + lr%32 ; LDS rows padded by 16B.
  for (int q = tid; q < 128 * 64; q += 256) {
    const int lr = q >> 6;
    const int ko = (q & 63) * 8;
    const int grow = (lr >> 5) * HH + w * 32 + (lr & 31);
    async_load_b128_to_lds(whh + (size_t)lr * 520 + ko,
                           Whh + (size_t)grow * HH + ko);
  }
  // zero syn slice and mem ping buffer 0 slice
  for (int e = tid; e < 512; e += 256) {
    syn[e] = 0.0f;
    memping[(e >> 5) * HH + w * 32 + (e & 31)] = 0;
  }
  wait_asynccnt0();
  __threadfence();
  grid_barrier(bar, 16 * 1);

  const int nbase = g * HH + w * 32 + c * 16;       // this wave's gate-column base in [0,4H)
  const unsigned short* whh_tile = whh + (size_t)wave * 16 * 520;

  for (int t = 0; t < SS; ++t) {
    const unsigned short* memin  = memping + (t & 1) * (16 * HH);
    unsigned short*       memout = memping + ((t + 1) & 1) * (16 * HH);
    f32x8 acc = {};
#pragma unroll 4
    for (int kk = 0; kk < HH / 32; ++kk) {
      acc = wmma_bf16(load_a(memin + kk * 32, HH, lane),
                      load_b(whh_tile + kk * 32, 520, lane), acc);
    }
    const unsigned short* gx = Gx + (size_t)t * 16 * FH + nbase;
    if (t + 1 < SS) __builtin_prefetch(Gx + (size_t)(t + 1) * 16 * FH + nbase, 0, 1);
#pragma unroll
    for (int j = 0; j < 8; ++j) {
      const int m = j + mb;
      gbuf[(g * 16 + m) * 32 + c * 16 + ncol] = acc[j] + bf2f(gx[(size_t)m * FH + ncol]);
    }
    __syncthreads();
    for (int e = tid; e < 512; e += 256) {
      const int bb = e >> 5, hh2 = e & 31;
      const float iv = sigf(gbuf[(0 * 16 + bb) * 32 + hh2]);
      const float fv = sigf(gbuf[(1 * 16 + bb) * 32 + hh2]);
      const float gv = tanhf(gbuf[(2 * 16 + bb) * 32 + hh2]);
      const float ov = sigf(gbuf[(3 * 16 + bb) * 32 + hh2]);
      const float sn = fv * syn[e] + iv * gv;
      const float mprev = bf2f(memin[bb * HH + w * 32 + hh2]);
      const float rst = (mprev > th) ? th : 0.0f;     // detached reset * threshold
      const float mn  = ov * tanhf(sn) - rst;
      syn[e] = sn;
      memout[bb * HH + w * 32 + hh2] = f2bf(mn);
      spks[((size_t)t * 16 + bb) * HH + w * 32 + hh2] = f2bf((mn > th) ? 1.0f : 0.0f);
    }
    __threadfence();
    grid_barrier(bar, 16 * (t + 2));
    __builtin_amdgcn_s_cluster_barrier();   // CDNA5 cluster path (NOP off-cluster)
  }
}

// ---- h = spks @ W_lin^T + b_lin (bf16 out) ------------------------------
__global__ __launch_bounds__(256) void k_lin(const unsigned short* __restrict__ spks,
                                             const unsigned short* __restrict__ Wlin,
                                             const float* __restrict__ b_lin,
                                             unsigned short* __restrict__ Hbuf) {
  const int lane = threadIdx.x & 31, wave = threadIdx.x >> 5;
  const int id = blockIdx.x * 8 + wave;
  const int mt = id >> 5;
  const int n  = (id & 31) * 16;
  const unsigned short* Abase = spks + (size_t)mt * 16 * HH;
  const unsigned short* Bbase = Wlin + (size_t)n * HH;
  f32x8 acc = {};
#pragma unroll 4
  for (int kk = 0; kk < HH / 32; ++kk)
    acc = wmma_bf16(load_a(Abase + kk * 32, HH, lane),
                    load_b(Bbase + kk * 32, HH, lane), acc);
  const int col = lane & 15, mb = (lane >= 16) ? 8 : 0;
  const float bias = b_lin[n + col];
#pragma unroll
  for (int j = 0; j < 8; ++j)
    Hbuf[((size_t)mt * 16 + j + mb) * HH + n + col] = f2bf(acc[j] + bias);
}

// ---- out = DyT( concat(h, x) @ W_mix^T + b_mix ) ------------------------
__global__ __launch_bounds__(256) void k_mix(const unsigned short* __restrict__ Hbuf,
                                             const unsigned short* __restrict__ xT,
                                             const unsigned short* __restrict__ Wmix,
                                             const float* __restrict__ b_mix,
                                             const float* __restrict__ alpha,
                                             const float* __restrict__ gamma,
                                             const float* __restrict__ beta,
                                             float* __restrict__ out) {
  const int lane = threadIdx.x & 31, wave = threadIdx.x >> 5;
  const int id = blockIdx.x * 8 + wave;
  const int mt = id >> 5;
  const int n  = (id & 31) * 16;
  const unsigned short* Bbase = Wmix + (size_t)n * (HH + II);
  f32x8 acc = {};
#pragma unroll 2
  for (int kk = 0; kk < HH / 32; ++kk)
    acc = wmma_bf16(load_a(Hbuf + (size_t)mt * 16 * HH + kk * 32, HH, lane),
                    load_b(Bbase + kk * 32, HH + II, lane), acc);
#pragma unroll 2
  for (int kk = 0; kk < II / 32; ++kk)
    acc = wmma_bf16(load_a(xT + (size_t)mt * 16 * II + kk * 32, II, lane),
                    load_b(Bbase + HH + kk * 32, HH + II, lane), acc);
  const int col = lane & 15, mb = (lane >= 16) ? 8 : 0;
  const float al = alpha[0];
  const float ga = gamma[n + col], be = beta[n + col], bm = b_mix[n + col];
#pragma unroll
  for (int j = 0; j < 8; ++j) {
    const int r = mt * 16 + j + mb;
    const int bb = r & 15;
    const int s  = r >> 4;
    out[((size_t)bb * SS + s) * HH + n + col] = ga * tanhf(al * (acc[j] + bm)) + be;
  }
}

extern "C" void kernel_launch(void* const* d_in, const int* in_sizes, int n_in,
                              void* d_out, int out_size, void* d_ws, size_t ws_size,
                              hipStream_t stream) {
  (void)in_sizes; (void)n_in; (void)out_size; (void)ws_size;
  const float* x     = (const float*)d_in[0];
  const float* W_ih  = (const float*)d_in[1];
  const float* W_hh  = (const float*)d_in[2];
  const float* b_ih  = (const float*)d_in[3];
  const float* b_hh  = (const float*)d_in[4];
  const float* thr   = (const float*)d_in[5];
  const float* W_lin = (const float*)d_in[6];
  const float* b_lin = (const float*)d_in[7];
  const float* W_mix = (const float*)d_in[8];
  const float* b_mix = (const float*)d_in[9];
  const float* alpha = (const float*)d_in[10];
  const float* gamma = (const float*)d_in[11];
  const float* beta  = (const float*)d_in[12];
  float* out = (float*)d_out;

  char* ws = (char*)d_ws;
  size_t off = 0;
  auto alloc = [&](size_t bytes) { char* p = ws + off; off += (bytes + 255) & ~(size_t)255; return p; };
  unsigned short* xT    = (unsigned short*)alloc((size_t)SS * 16 * II * 2);
  unsigned short* Gx    = (unsigned short*)alloc((size_t)SS * 16 * FH * 2);
  unsigned short* Wihb  = (unsigned short*)alloc((size_t)FH * II * 2);
  unsigned short* Whhb  = (unsigned short*)alloc((size_t)FH * HH * 2);
  unsigned short* Wlinb = (unsigned short*)alloc((size_t)HH * HH * 2);
  unsigned short* Wmixb = (unsigned short*)alloc((size_t)HH * (HH + II) * 2);
  unsigned short* spks  = (unsigned short*)alloc((size_t)SS * 16 * HH * 2);
  unsigned short* Hbuf  = (unsigned short*)alloc((size_t)SS * 16 * HH * 2);
  unsigned short* memp  = (unsigned short*)alloc((size_t)2 * 16 * HH * 2);
  int* bar = (int*)alloc(256);

  hipMemsetAsync(bar, 0, sizeof(int), stream);
  k_conv_xT<<<4096, 256, 0, stream>>>(x, xT);
  k_f32_to_bf16<<<1024, 256, 0, stream>>>(W_ih, Wihb, FH * II);
  k_f32_to_bf16<<<1024, 256, 0, stream>>>(W_hh, Whhb, FH * HH);
  k_f32_to_bf16<<<512, 256, 0, stream>>>(W_lin, Wlinb, HH * HH);
  k_f32_to_bf16<<<512, 256, 0, stream>>>(W_mix, Wmixb, HH * (HH + II));
  k_gemm_x<<<SS * 16, 256, 0, stream>>>(xT, Wihb, b_ih, b_hh, Gx);
  const size_t smem = 128 * 520 * sizeof(unsigned short) + (4 * 16 * 32 + 512) * sizeof(float);
  k_scan<<<16, 256, smem, stream>>>(Gx, Whhb, thr, spks, memp, bar);
  k_lin<<<8192, 256, 0, stream>>>(spks, Wlinb, b_lin, Hbuf);
  k_mix<<<8192, 256, 0, stream>>>(Hbuf, xT, Wmixb, b_mix, alpha, gamma, beta, out);
}
